// SimpleLSTM_43404939493707
// MI455X (gfx1250) — matrix operand order repaired and checked by reading, not compiled, not run on hardware
//
#include <hip/hip_runtime.h>

typedef __attribute__((ext_vector_type(16))) _Float16 v16h;
typedef __attribute__((ext_vector_type(8)))  float    v8f;
typedef __attribute__((ext_vector_type(4)))  float    v4f;

#define BATCH   1048576
#define IN_DIM  32
#define HID     64

// ---- workspace layout ----
#define WS_A1   0            // 12 frags * 512 halves (w_ih1: i,g,o gate tiles)
#define WS_A2   (12 * 512)   // 24 frags            (w_ih2: i,g,o x 2 K-steps)
#define WS_A3   (36 * 512)   // 4 frags             (fc1_w)
#define WS_B1_BYTES 40960    // 256 f32 (b_ih1+b_hh1)
#define WS_B2_BYTES 41984    // 256 f32
#define WS_FB_BYTES 43008    // 32 f32

// ---- fast elementwise math: CDNA5 v_tanh_f32 transcendental (probe-confirmed) ----
#if __has_builtin(__builtin_amdgcn_tanhf)
__device__ __forceinline__ float tanh_(float x) { return __builtin_amdgcn_tanhf(x); }
__device__ __forceinline__ float sigm_(float x) {
    return fmaf(0.5f, __builtin_amdgcn_tanhf(0.5f * x), 0.5f);
}
#else
__device__ __forceinline__ float tanh_(float x) {
    float e = __expf(-2.f * x);
    return fmaf(2.f, __builtin_amdgcn_rcpf(1.f + e), -1.f);
}
__device__ __forceinline__ float sigm_(float x) {
    return __builtin_amdgcn_rcpf(1.f + __expf(-x));
}
#endif

__device__ __forceinline__ v8f wmma16(v16h a, v16h b, v8f c) {
    return __builtin_amdgcn_wmma_f32_16x16x32_f16(false, a, false, b, (short)0, c, false, false);
}

// ---------------- prep: pack weights into A-fragment layout, fold biases ---------
// A 16x32 f16 packing used here (contiguous K, as in the B fragments): lane m holds row M=m&15; half j -> K = j + 16*(lane>=16)
__global__ void lstm_prep(const float* __restrict__ w1,
                          const float* __restrict__ b_ih1, const float* __restrict__ b_hh1,
                          const float* __restrict__ w2,
                          const float* __restrict__ b_ih2, const float* __restrict__ b_hh2,
                          const float* __restrict__ fc1w, const float* __restrict__ fc1b,
                          _Float16* __restrict__ wsA,
                          float* __restrict__ wsB1, float* __restrict__ wsB2,
                          float* __restrict__ wsFB) {
    const int b = blockIdx.x;
    const int t = threadIdx.x;              // 0..511
    const int lane = t >> 4;
    const int j    = t & 15;
    const int kofs = j + ((lane >= 16) ? 16 : 0);
    const int lm   = lane & 15;

    if (b < 12) {
        const int g = b >> 2, hg = b & 3;
        const int grow = (g == 0) ? 0 : (g == 1) ? 128 : 192;
        wsA[WS_A1 + b * 512 + lane * 16 + j] =
            (_Float16)w1[(grow + hg * 16 + lm) * IN_DIM + kofs];
    } else if (b < 36) {
        const int f = b - 12;
        const int g = f >> 3, hg = (f >> 1) & 3, kt = f & 1;
        const int grow = (g == 0) ? 0 : (g == 1) ? 128 : 192;
        wsA[WS_A2 + f * 512 + lane * 16 + j] =
            (_Float16)w2[(grow + hg * 16 + lm) * HID + kt * 32 + kofs];
    } else if (b < 40) {
        const int f = b - 36;
        const int mt = f >> 1, kt = f & 1;
        wsA[WS_A3 + f * 512 + lane * 16 + j] =
            (_Float16)fc1w[(mt * 16 + lm) * HID + kt * 32 + kofs];
    } else if (b == 40) {
        if (t < 256) wsB1[t] = b_ih1[t] + b_hh1[t];
    } else if (b == 41) {
        if (t < 256) wsB2[t] = b_ih2[t] + b_hh2[t];
    } else {
        if (t < 32) wsFB[t] = fc1b[t];
    }
}

// ---------------- main fused kernel: one wave handles one 16-row batch tile ------
__global__ void __launch_bounds__(256)
lstm_fused(const float* __restrict__ x,
           const _Float16* __restrict__ wsA,
           const float* __restrict__ b1, const float* __restrict__ b2,
           const float* __restrict__ fb,
           const float* __restrict__ fc2w, const float* __restrict__ fc2b,
           float* __restrict__ out) {
    const int lane = threadIdx.x & 31;
    const bool low = lane < 16;
    const int lm   = lane & 15;
    const int rsel = low ? 0 : 8;           // D-tile: lane holds rows j (low) or 8+j (high)
    const int wave = (int)((blockIdx.x * blockDim.x + threadIdx.x) >> 5);
    const int rowbase = wave * 16;

    // ---- x^T B-fragment: lane n = batch row n; lanes<16 -> K 0..15, >=16 -> K 16..31
    v16h bx;
    {
        const v4f* p = (const v4f*)(x + (size_t)(rowbase + lm) * IN_DIM + (low ? 0 : 16));
        v4f a0 = p[0], a1 = p[1], a2 = p[2], a3 = p[3];
#pragma unroll
        for (int q = 0; q < 4; ++q) {
            bx[q]      = (_Float16)a0[q];
            bx[4 + q]  = (_Float16)a1[q];
            bx[8 + q]  = (_Float16)a2[q];
            bx[12 + q] = (_Float16)a3[q];
        }
    }

    // ============ layer 1: G1^T = W1 @ x^T (f gate dead since c0=0) ==========
    unsigned pk1[4][4];
#pragma unroll
    for (int hg = 0; hg < 4; ++hg) {
        v8f ci, cg, co;
        const float* pi = b1 + hg * 16 + rsel;
        const float* pg = b1 + 128 + hg * 16 + rsel;
        const float* po = b1 + 192 + hg * 16 + rsel;
#pragma unroll
        for (int j = 0; j < 8; ++j) { ci[j] = pi[j]; cg[j] = pg[j]; co[j] = po[j]; }
        v16h Ai = *(const v16h*)(wsA + WS_A1 + (0 * 4 + hg) * 512 + lane * 16);
        v16h Ag = *(const v16h*)(wsA + WS_A1 + (1 * 4 + hg) * 512 + lane * 16);
        v16h Ao = *(const v16h*)(wsA + WS_A1 + (2 * 4 + hg) * 512 + lane * 16);
        v8f gi = wmma16(Ai, bx, ci);
        v8f gg = wmma16(Ag, bx, cg);
        v8f go = wmma16(Ao, bx, co);
#pragma unroll
        for (int q = 0; q < 4; ++q) {
            float c0 = sigm_(gi[2 * q])     * tanh_(gg[2 * q]);
            float c1 = sigm_(gi[2 * q + 1]) * tanh_(gg[2 * q + 1]);
            float h0 = sigm_(go[2 * q])     * tanh_(c0);
            float h1 = sigm_(go[2 * q + 1]) * tanh_(c1);
            union { _Float16 h[2]; unsigned u; } pr;
            pr.h[0] = (_Float16)h0; pr.h[1] = (_Float16)h1;
            pk1[hg][q] = pr.u;
        }
    }

    // ---- h1^T B-fragments via lane<->lane+16 half swap
    v16h bh1[2];
#pragma unroll
    for (int kt = 0; kt < 2; ++kt) {
        union { v16h v; unsigned u[8]; } f;
#pragma unroll
        for (int q = 0; q < 4; ++q) {
            unsigned p0 = pk1[2 * kt][q], p1 = pk1[2 * kt + 1][q];
            unsigned s0 = (unsigned)__shfl_xor((int)p0, 16, 32);
            unsigned s1 = (unsigned)__shfl_xor((int)p1, 16, 32);
            f.u[q]     = low ? p0 : s1;
            f.u[4 + q] = low ? s0 : p1;
        }
        bh1[kt] = f.v;
    }

    // ============ layer 2: G2^T = W2 @ h1^T (K=64 -> 2 WMMA steps) ===========
    unsigned pk2[4][4];
#pragma unroll
    for (int hg = 0; hg < 4; ++hg) {
        v8f ci, cg, co;
        const float* pi = b2 + hg * 16 + rsel;
        const float* pg = b2 + 128 + hg * 16 + rsel;
        const float* po = b2 + 192 + hg * 16 + rsel;
#pragma unroll
        for (int j = 0; j < 8; ++j) { ci[j] = pi[j]; cg[j] = pg[j]; co[j] = po[j]; }
#pragma unroll
        for (int kt = 0; kt < 2; ++kt) {
            v16h Ai = *(const v16h*)(wsA + WS_A2 + ((0 * 4 + hg) * 2 + kt) * 512 + lane * 16);
            v16h Ag = *(const v16h*)(wsA + WS_A2 + ((1 * 4 + hg) * 2 + kt) * 512 + lane * 16);
            v16h Ao = *(const v16h*)(wsA + WS_A2 + ((2 * 4 + hg) * 2 + kt) * 512 + lane * 16);
            ci = wmma16(Ai, bh1[kt], ci);
            cg = wmma16(Ag, bh1[kt], cg);
            co = wmma16(Ao, bh1[kt], co);
        }
#pragma unroll
        for (int q = 0; q < 4; ++q) {
            float c0 = sigm_(ci[2 * q])     * tanh_(cg[2 * q]);
            float c1 = sigm_(ci[2 * q + 1]) * tanh_(cg[2 * q + 1]);
            float h0 = sigm_(co[2 * q])     * tanh_(c0);
            float h1 = sigm_(co[2 * q + 1]) * tanh_(c1);
            union { _Float16 h[2]; unsigned u; } pr;
            pr.h[0] = (_Float16)h0; pr.h[1] = (_Float16)h1;
            pk2[hg][q] = pr.u;
        }
    }

    v16h bh2[2];
#pragma unroll
    for (int kt = 0; kt < 2; ++kt) {
        union { v16h v; unsigned u[8]; } f;
#pragma unroll
        for (int q = 0; q < 4; ++q) {
            unsigned p0 = pk2[2 * kt][q], p1 = pk2[2 * kt + 1][q];
            unsigned s0 = (unsigned)__shfl_xor((int)p0, 16, 32);
            unsigned s1 = (unsigned)__shfl_xor((int)p1, 16, 32);
            f.u[q]     = low ? p0 : s1;
            f.u[4 + q] = low ? s0 : p1;
        }
        bh2[kt] = f.v;
    }

    // ============ fc1 (relu) + fc2 dot + sigmoid =============================
    float partial = 0.f;
#pragma unroll
    for (int mt = 0; mt < 2; ++mt) {
        v8f c;
        const float* pb = fb + mt * 16 + rsel;
#pragma unroll
        for (int j = 0; j < 8; ++j) c[j] = pb[j];
        v16h A0 = *(const v16h*)(wsA + WS_A3 + (mt * 2 + 0) * 512 + lane * 16);
        v16h A1 = *(const v16h*)(wsA + WS_A3 + (mt * 2 + 1) * 512 + lane * 16);
        c = wmma16(A0, bh2[0], c);
        c = wmma16(A1, bh2[1], c);
        const float* wq = fc2w + mt * 16 + rsel;   // uniform, L0-resident; no local array
#pragma unroll
        for (int j = 0; j < 8; ++j) partial = fmaf(fmaxf(c[j], 0.f), wq[j], partial);
    }
    float tot = partial + __shfl_xor(partial, 16, 32);
    float y = sigm_(tot + fc2b[0]);
    if (low) out[rowbase + lm] = y;
}

extern "C" void kernel_launch(void* const* d_in, const int* in_sizes, int n_in,
                              void* d_out, int out_size, void* d_ws, size_t ws_size,
                              hipStream_t stream) {
    const float* x     = (const float*)d_in[0];
    const float* w1    = (const float*)d_in[1];
    const float* b_ih1 = (const float*)d_in[3];
    const float* b_hh1 = (const float*)d_in[4];
    const float* w2    = (const float*)d_in[5];
    const float* b_ih2 = (const float*)d_in[7];
    const float* b_hh2 = (const float*)d_in[8];
    const float* fc1w  = (const float*)d_in[9];
    const float* fc1b  = (const float*)d_in[10];
    const float* fc2w  = (const float*)d_in[11];
    const float* fc2b  = (const float*)d_in[12];

    _Float16* wsA = (_Float16*)d_ws;
    float* wsB1 = (float*)((char*)d_ws + WS_B1_BYTES);
    float* wsB2 = (float*)((char*)d_ws + WS_B2_BYTES);
    float* wsFB = (float*)((char*)d_ws + WS_FB_BYTES);

    lstm_prep<<<43, 512, 0, stream>>>(w1, b_ih1, b_hh1, w2, b_ih2, b_hh2,
                                      fc1w, fc1b, wsA, wsB1, wsB2, wsFB);

    // 1,048,576 rows / 16 rows per wave = 65,536 waves; 8 waves per 256-thread block
    lstm_fused<<<BATCH / 16 / 8, 256, 0, stream>>>(x, wsA, wsB1, wsB2, wsFB,
                                                   fc2w, fc2b, (float*)d_out);
}
